// TriAANBlock_42949673528
// MI455X (gfx1250) — matrix operand order repaired and checked by reading, not compiled, not run on hardware
//
#include <hip/hip_runtime.h>
#include <stdint.h>

// Problem constants (reference: B=4, C=512, Ts=Tt=S=2048)
#define BB 4
#define CC 512
#define TT 2048
#define SS 2048
#define EPS 1e-5f

typedef __attribute__((ext_vector_type(16))) _Float16 v16h;
typedef __attribute__((ext_vector_type(8)))  float    v8f;

__device__ __forceinline__ v8f wmma16(v16h a, v16h b, v8f c) {
  // D = A(16x32 f16) * B(32x16 f16) + C(16x16 f32)
  return __builtin_amdgcn_wmma_f32_16x16x32_f16(false, a, false, b, (short)0, c,
                                                false, false);
}

union U16h { v16h h; uint32_t u[8]; };

// ---------------------------------------------------------------------------
// CDNA5 async global->LDS DMA (GLOBAL_LOAD_ASYNC_TO_LDS_B128, ASYNCcnt).
// lds_off = workgroup-relative LDS byte address (low 32 bits of flat address),
// gaddr   = 64-bit global address in a VGPR pair, GV addressing mode.
// ---------------------------------------------------------------------------
__device__ __forceinline__ void async_g2l_b128(uint32_t lds_off, const void* gptr) {
  asm volatile("global_load_async_to_lds_b128 %0, %1, off"
               :: "v"(lds_off), "v"((uint64_t)(size_t)gptr)
               : "memory");
}
__device__ __forceinline__ void wait_async() {
  asm volatile("s_wait_asynccnt 0" ::: "memory");
}

// ---------------------------------------------------------------------------
// LDS tiles: row stride 20 u32 (40 halfs). Each 32-half row = 4 x 16B chunks,
// 16B-aligned (row*80B + ch*16B), and 20*r mod 64 banks is conflict-free for
// the 16 row-readers of a fragment.
// A-fragment layout: lane L: row = L%16; half i -> K = i + (L>=16?16:0)
//   => u32 pairs at {0..7} (+8 u32 for hi lanes)
// B-fragment layout (same k order as A): lane L: col n = L%16; half i -> K = i + (L>=16?16:0)
// ---------------------------------------------------------------------------
#define LSTR 20
__device__ __forceinline__ v16h fragA_lds(const uint32_t* As32, int mlocal, int hi) {
  U16h f;
  int base = mlocal * LSTR + (hi ? 8 : 0);
  f.u[0] = As32[base + 0];  f.u[1] = As32[base + 1];
  f.u[2] = As32[base + 2];  f.u[3] = As32[base + 3];
  f.u[4] = As32[base + 4];  f.u[5] = As32[base + 5];
  f.u[6] = As32[base + 6]; f.u[7] = As32[base + 7];
  return f.h;
}
__device__ __forceinline__ v16h fragB_lds(const uint32_t* Bs32, int nlocal, int hi) {
  U16h f;
  int base = nlocal * LSTR + (hi ? 8 : 0);
#pragma unroll
  for (int j = 0; j < 8; ++j) f.u[j] = Bs32[base + j];
  return f.h;
}
// A fragment straight from global row-major memory (16B-aligned chunks).
__device__ __forceinline__ v16h fragA_g(const _Float16* row, int hi) {
  const uint32_t* p = (const uint32_t*)(row + (hi ? 16 : 0));
  U16h f;
  f.u[0] = p[0]; f.u[1] = p[1]; f.u[2] = p[2];  f.u[3] = p[3];
  f.u[4] = p[4]; f.u[5] = p[5]; f.u[6] = p[6]; f.u[7] = p[7];
  return f.h;
}

// ---------------------------------------------------------------------------
// small elementwise / reduction kernels
// ---------------------------------------------------------------------------
__global__ void k_f32f16(const float* __restrict__ in, _Float16* __restrict__ out,
                         int n, float scale) {
  int i = blockIdx.x * 256 + threadIdx.x;
  if (i < n) out[i] = (_Float16)(in[i] * scale);
}

// conv2_w [O][C][3] -> W3h [3][O][C] f16
__global__ void k_w3(const float* __restrict__ w, _Float16* __restrict__ out) {
  int i = blockIdx.x * 256 + threadIdx.x;  // o*C + c
  if (i < CC * CC) {
#pragma unroll
    for (int d = 0; d < 3; ++d) out[d * CC * CC + i] = (_Float16)w[i * 3 + d];
  }
}

__global__ void k_bn(const float* __restrict__ g, const float* __restrict__ b,
                     const float* __restrict__ m, const float* __restrict__ v,
                     float* __restrict__ sc, float* __restrict__ sh) {
  int o = blockIdx.x * 256 + threadIdx.x;
  if (o < CC) {
    float s = g[o] * rsqrtf(v[o] + EPS);
    sc[o] = s;
    sh[o] = b[o] - m[o] * s;
  }
}

// stats over channel axis per (b,t): input [B,C,T]  (tinorm, ddof=1)
__global__ void k_tstats(const float* __restrict__ in, float* __restrict__ om,
                         float* __restrict__ osv) {
  int b = blockIdx.y;
  int t = blockIdx.x * 64 + threadIdx.x;  // block (64,4)
  const float* p = in + (size_t)b * CC * TT + t;
  float s = 0.f, q = 0.f;
  for (int c = threadIdx.y; c < CC; c += 4) {
    float v = p[(size_t)c * TT];
    s += v; q += v * v;
  }
  __shared__ float rs[4][64], rq[4][64];
  rs[threadIdx.y][threadIdx.x] = s;
  rq[threadIdx.y][threadIdx.x] = q;
  __syncthreads();
  if (threadIdx.y == 0) {
    s = rs[0][threadIdx.x] + rs[1][threadIdx.x] + rs[2][threadIdx.x] + rs[3][threadIdx.x];
    q = rq[0][threadIdx.x] + rq[1][threadIdx.x] + rq[2][threadIdx.x] + rq[3][threadIdx.x];
    float mean = s / CC;
    float var = fmaxf((q - s * mean) / (CC - 1), 0.f);
    om[b * TT + t] = mean;
    osv[b * TT + t] = 1.f / (sqrtf(var) + EPS);
  }
}

// generic strided stats per (b,c) over nt elements (inorm, ddof=1)
__global__ void k_sstats(const float* __restrict__ in, float* __restrict__ om,
                         float* __restrict__ osv, long long bstride, long long cstride,
                         long long tstride, int nc, int nt) {
  int c = blockIdx.x * 256 + threadIdx.x;
  int b = blockIdx.y;
  if (c >= nc) return;
  const float* p = in + (size_t)b * bstride + (size_t)c * cstride;
  float s = 0.f, q = 0.f;
  for (int t = 0; t < nt; ++t) {
    float v = p[(size_t)t * tstride];
    s += v; q += v * v;
  }
  float mean = s / nt;
  float var = fmaxf((q - s * mean) / (nt - 1), 0.f);
  om[b * nc + c] = mean;
  osv[b * nc + c] = 1.f / (sqrtf(var) + EPS);
}

// [B,C,T] f32 -> transposed normalized f16 [B,T,C]: Xt (tinorm), Xi (inorm), Yr raw
__global__ void k_normxpose(const float* __restrict__ in, const float* __restrict__ tm,
                            const float* __restrict__ tsv, const float* __restrict__ im,
                            const float* __restrict__ isv, _Float16* __restrict__ Xt,
                            _Float16* __restrict__ Xi, _Float16* __restrict__ Yr) {
  int b = blockIdx.z;
  int t0 = blockIdx.x * 32, c0 = blockIdx.y * 32;
  __shared__ float tile[32][33];
#pragma unroll
  for (int j = 0; j < 4; ++j) {
    int c = c0 + threadIdx.y + 8 * j;
    tile[threadIdx.y + 8 * j][threadIdx.x] =
        in[((size_t)b * CC + c) * TT + t0 + threadIdx.x];
  }
  __syncthreads();
#pragma unroll
  for (int j = 0; j < 4; ++j) {
    int t = t0 + threadIdx.y + 8 * j;
    int c = c0 + threadIdx.x;
    float v = tile[threadIdx.x][threadIdx.y + 8 * j];
    size_t o = ((size_t)b * TT + t) * CC + c;
    Xt[o] = (_Float16)((v - tm[b * TT + t]) * tsv[b * TT + t]);
    Xi[o] = (_Float16)((v - im[b * CC + c]) * isv[b * CC + c]);
    if (Yr) Yr[o] = (_Float16)v;
  }
}

// f16 [B,T,C] -> f16 [B,C,T]
__global__ void k_xpose16(const _Float16* __restrict__ in, _Float16* __restrict__ out) {
  int b = blockIdx.z;
  int t0 = blockIdx.x * 32, c0 = blockIdx.y * 32;
  __shared__ _Float16 tile[32][34];
#pragma unroll
  for (int j = 0; j < 4; ++j) {
    int t = t0 + threadIdx.y + 8 * j;
    tile[threadIdx.y + 8 * j][threadIdx.x] =
        in[((size_t)b * TT + t) * CC + c0 + threadIdx.x];
  }
  __syncthreads();
#pragma unroll
  for (int j = 0; j < 4; ++j) {
    int c = c0 + threadIdx.y + 8 * j;
    out[((size_t)b * CC + c) * TT + t0 + threadIdx.x] =
        tile[threadIdx.x][threadIdx.y + 8 * j];
  }
}

// ---------------------------------------------------------------------------
// WMMA GEMM: C[M,N](f16) = A[M,K](f16) @ W[N,K]^T + bias ; 64x128 block, 8 waves.
// Tiles staged into LDS via async DMA (global_load_async_to_lds_b128).
// ---------------------------------------------------------------------------
__global__ __launch_bounds__(256) void k_gemm(
    const _Float16* __restrict__ A, long long strideA, int lda,
    const _Float16* __restrict__ W, long long strideW,
    const float* __restrict__ bias, long long strideBias,
    _Float16* __restrict__ Cout, long long strideC, int ldc, int K) {
  __shared__ uint32_t As32[64 * LSTR];
  __shared__ uint32_t Bs32[128 * LSTR];
  int b = blockIdx.z;
  A += (size_t)b * strideA;
  W += (size_t)b * strideW;
  Cout += (size_t)b * strideC;
  int m0 = blockIdx.x * 64, n0 = blockIdx.y * 128;
  int tid = threadIdx.x;
  int lane = tid & 31, wv = tid >> 5;
  int ln = lane & 15, hi = lane >> 4;
  int mi = wv & 3, ng = wv >> 2;
  int rowA = tid >> 2, chA = tid & 3;  // 64 rows x 4 chunks of 16B
  v8f acc[4] = {};
  for (int kb = 0; kb < K; kb += 32) {
    async_g2l_b128((uint32_t)(size_t)(As32 + rowA * LSTR + chA * 4),
                   A + (size_t)(m0 + rowA) * lda + kb + chA * 8);
#pragma unroll
    for (int i = 0; i < 2; ++i) {  // 128 rows x 4 chunks
      int idx = tid + 256 * i;
      int rowB = idx >> 2, chB = idx & 3;
      async_g2l_b128((uint32_t)(size_t)(Bs32 + rowB * LSTR + chB * 4),
                     W + (size_t)(n0 + rowB) * K + kb + chB * 8);
    }
    wait_async();
    __syncthreads();
    v16h af = fragA_lds(As32, mi * 16 + ln, hi);
#pragma unroll
    for (int tt = 0; tt < 4; ++tt) {
      v16h bf = fragB_lds(Bs32, (ng * 4 + tt) * 16 + ln, hi);
      acc[tt] = wmma16(af, bf, acc[tt]);
    }
    __syncthreads();
  }
#pragma unroll
  for (int tt = 0; tt < 4; ++tt) {
    int n = n0 + (ng * 4 + tt) * 16 + ln;
    float bvv = bias ? bias[(size_t)b * strideBias + n] : 0.f;
#pragma unroll
    for (int r = 0; r < 8; ++r) {
      int m = m0 + mi * 16 + r + (hi ? 8 : 0);
      Cout[(size_t)m * ldc + n] = (_Float16)(acc[tt][r] + bvv);
    }
  }
}

// conv2 (k=3, pad 1) + BN(eval) + ReLU, A = zero-padded y1 [B,T+2,C], out f32 [B,T,C]
__global__ __launch_bounds__(256) void k_conv2(
    const _Float16* __restrict__ Y1, const _Float16* __restrict__ W3,
    const float* __restrict__ bnsc, const float* __restrict__ bnsh,
    float* __restrict__ X) {
  __shared__ uint32_t As32[64 * LSTR];
  __shared__ uint32_t Bs32[128 * LSTR];
  int b = blockIdx.z;
  const _Float16* A = Y1 + (size_t)b * (TT + 2) * CC;
  int m0 = blockIdx.x * 64, n0 = blockIdx.y * 128;
  int tid = threadIdx.x;
  int lane = tid & 31, wv = tid >> 5;
  int ln = lane & 15, hi = lane >> 4;
  int mi = wv & 3, ng = wv >> 2;
  int rowA = tid >> 2, chA = tid & 3;
  v8f acc[4] = {};
  for (int d = 0; d < 3; ++d) {
    const _Float16* Wd = W3 + (size_t)d * CC * CC;
    for (int kb = 0; kb < CC; kb += 32) {
      async_g2l_b128((uint32_t)(size_t)(As32 + rowA * LSTR + chA * 4),
                     A + (size_t)(m0 + rowA + d) * CC + kb + chA * 8);
#pragma unroll
      for (int i = 0; i < 2; ++i) {
        int idx = tid + 256 * i;
        int rowB = idx >> 2, chB = idx & 3;
        async_g2l_b128((uint32_t)(size_t)(Bs32 + rowB * LSTR + chB * 4),
                       Wd + (size_t)(n0 + rowB) * CC + kb + chB * 8);
      }
      wait_async();
      __syncthreads();
      v16h af = fragA_lds(As32, mi * 16 + ln, hi);
#pragma unroll
      for (int tt = 0; tt < 4; ++tt) {
        v16h bf = fragB_lds(Bs32, (ng * 4 + tt) * 16 + ln, hi);
        acc[tt] = wmma16(af, bf, acc[tt]);
      }
      __syncthreads();
    }
  }
#pragma unroll
  for (int tt = 0; tt < 4; ++tt) {
    int n = n0 + (ng * 4 + tt) * 16 + ln;
    float sc = bnsc[n], sh = bnsh[n];
#pragma unroll
    for (int r = 0; r < 8; ++r) {
      int m = m0 + mi * 16 + r + (hi ? 8 : 0);
      X[((size_t)b * TT + m) * CC + n] = fmaxf(acc[tt][r] * sc + sh, 0.f);
    }
  }
}

// ---------------------------------------------------------------------------
// Flash-attention with fused first/second moment column reduction.
// Q,K: [B,T,C] f16 (q pre-scaled by 1/sqrt(C)); Vt: [B,C,T] f16.
// Accumulates sum_s attn-mean and sum_s relu(E[v^2]-mean^2) into Gmean/Gvar [B,C].
// Block: 32 queries, 256 threads (8 waves). Wave w: S-tile (mi=w>>2, nj=w&3);
// accumulation tiles (mi, c-tiles (w&3)+4j). K/V are L2-resident (2MB/batch).
// ---------------------------------------------------------------------------
__global__ __launch_bounds__(256) void k_attn(const _Float16* __restrict__ Q,
                                              const _Float16* __restrict__ Km,
                                              const _Float16* __restrict__ Vt,
                                              float* __restrict__ Gmean,
                                              float* __restrict__ Gvar) {
  int b = blockIdx.y;
  int q0 = blockIdx.x * 32;
  const _Float16* Qb = Q + (size_t)b * TT * CC;
  const _Float16* Kb = Km + (size_t)b * TT * CC;
  const _Float16* Vb = Vt + (size_t)b * CC * TT;
  int tid = threadIdx.x, lane = tid & 31, wv = tid >> 5;
  int ln = lane & 15, hi = lane >> 4;
  int mi = wv >> 2;
  int nj = wv & 3;
  int cset = wv & 3;

  __shared__ float Sb[32][65];
  __shared__ __align__(16) _Float16 Pb[32][64];
  __shared__ float mrow[32], lrow[32], arow[32];
  __shared__ float red[32][8];
  __shared__ float colm[CC], colv[CC];

  for (int i = tid; i < 32; i += 256) { mrow[i] = -3.0e38f; lrow[i] = 0.f; }
  for (int i = tid; i < CC; i += 256) { colm[i] = 0.f; colv[i] = 0.f; }

  // register-resident Q fragments for this wave's 16-row tile (full K=512)
  v16h qf[16];
  {
    const _Float16* qp = Qb + (size_t)(q0 + mi * 16 + ln) * CC;
#pragma unroll
    for (int i = 0; i < 16; ++i) qf[i] = fragA_g(qp + i * 32, hi);
  }
  v8f am[8] = {}, a2[8] = {};
  __syncthreads();

  for (int j0 = 0; j0 < TT; j0 += 64) {
    // prefetch next key/value block into L0/L2 (global_prefetch_b8)
    if (j0 + 64 < TT) {
      __builtin_prefetch(Kb + (size_t)(j0 + 64 + nj * 16 + ln) * CC, 0, 0);
      __builtin_prefetch(Vb + (size_t)(cset * 16 + ln) * TT + j0 + 64, 0, 0);
    }
    // ---- S = Q K^T (logits), one 16x16 tile per wave
    v8f s = {};
    {
      const _Float16* kp = Kb + (size_t)(j0 + nj * 16 + ln) * CC + (hi ? 16 : 0);
#pragma unroll
      for (int i = 0; i < 16; ++i) {
        v16h bf = *(const v16h*)(kp + i * 32);
        s = wmma16(qf[i], bf, s);
      }
    }
#pragma unroll
    for (int r = 0; r < 8; ++r)
      Sb[mi * 16 + r + (hi ? 8 : 0)][nj * 16 + ln] = s[r];
    __syncthreads();

    // ---- online softmax over this 64-key block
    int row = tid & 31, grp = tid >> 5;
    float pm = -3.0e38f;
#pragma unroll
    for (int c = 0; c < 8; ++c) pm = fmaxf(pm, Sb[row][grp * 8 + c]);
    red[row][grp] = pm;
    __syncthreads();
    if (grp == 0) {
      float bm = red[row][0];
#pragma unroll
      for (int g = 1; g < 8; ++g) bm = fmaxf(bm, red[row][g]);
      float mold = mrow[row];
      float mnew = fmaxf(mold, bm);
      float al = __expf(mold - mnew);
      arow[row] = al;
      mrow[row] = mnew;
      lrow[row] *= al;
    }
    __syncthreads();
    {
      float mn = mrow[row];
      float ps = 0.f;
#pragma unroll
      for (int c = 0; c < 8; ++c) {
        float pv = __expf(Sb[row][grp * 8 + c] - mn);
        Pb[row][grp * 8 + c] = (_Float16)pv;
        ps += pv;
      }
      red[row][grp] = ps;
    }
    __syncthreads();
    if (grp == 0) {
      float tsum = 0.f;
#pragma unroll
      for (int g = 0; g < 8; ++g) tsum += red[row][g];
      lrow[row] += tsum;
    }
    __syncthreads();

    // ---- rescale running accumulators by alpha(row)
    float av[8];
#pragma unroll
    for (int r = 0; r < 8; ++r) av[r] = arow[mi * 16 + r + (hi ? 8 : 0)];
#pragma unroll
    for (int ct = 0; ct < 8; ++ct)
#pragma unroll
      for (int r = 0; r < 8; ++r) { am[ct][r] *= av[r]; a2[ct][r] *= av[r]; }

    // ---- accumulate P@V and P@(V*V)
    v16h pf[2];
    {
      const uint32_t* P32 = (const uint32_t*)&Pb[0][0];
      int m = mi * 16 + ln;
#pragma unroll
      for (int ks = 0; ks < 2; ++ks) {
        int base = m * 32 + ks * 16 + (hi ? 8 : 0);
        U16h f;
        f.u[0] = P32[base + 0]; f.u[1] = P32[base + 1];
        f.u[2] = P32[base + 2]; f.u[3] = P32[base + 3];
        f.u[4] = P32[base + 4]; f.u[5] = P32[base + 5];
        f.u[6] = P32[base + 6]; f.u[7] = P32[base + 7];
        pf[ks] = f.h;
      }
    }
#pragma unroll
    for (int ct = 0; ct < 8; ++ct) {
      int c0 = (cset + 4 * ct) * 16;
      const _Float16* vp = Vb + (size_t)(c0 + ln) * TT + j0 + (hi ? 16 : 0);
#pragma unroll
      for (int ks = 0; ks < 2; ++ks) {
        v16h bvv = *(const v16h*)(vp + ks * 32);
        am[ct] = wmma16(pf[ks], bvv, am[ct]);
        v16h b2 = bvv * bvv;
        a2[ct] = wmma16(pf[ks], b2, a2[ct]);
      }
    }
    __syncthreads();
  }

  // ---- finalize: divide by softmax denom, var = relu(m2-mean^2), column-sum
  float linv[8];
#pragma unroll
  for (int r = 0; r < 8; ++r) linv[r] = 1.f / lrow[mi * 16 + r + (hi ? 8 : 0)];
#pragma unroll
  for (int ct = 0; ct < 8; ++ct) {
    int c = (cset + 4 * ct) * 16 + ln;
    float sm = 0.f, sv = 0.f;
#pragma unroll
    for (int r = 0; r < 8; ++r) {
      float me = am[ct][r] * linv[r];
      float m2 = a2[ct][r] * linv[r];
      sm += me;
      sv += fmaxf(m2 - me * me, 0.f);
    }
    sm += __shfl_xor(sm, 16);
    sv += __shfl_xor(sv, 16);
    if (hi == 0) {
      atomicAdd(&colm[c], sm);
      atomicAdd(&colv[c], sv);
    }
  }
  __syncthreads();
  for (int i = tid; i < CC; i += 256) {
    atomicAdd(&Gmean[b * CC + i], colm[i]);
    atomicAdd(&Gvar[b * CC + i], colv[i]);
  }
}

// Gmean -> mean_c ; Gvar -> std_c = sqrt(mean var)
__global__ void k_statfin(float* m, float* v, int n, float invT) {
  int i = blockIdx.x * 256 + threadIdx.x;
  if (i < n) {
    m[i] *= invT;
    v[i] = sqrtf(fmaxf(v[i] * invT, 0.f));
  }
}

// Weff[b][o][c] = w1[o][c]*stdA[b][c] + w2[o][c]*stdB[b][c]  (conv1 folded)
// beff[b][o]   = sum_c(w1*meanA + w2*meanB) + conv1_b[o]
__global__ void k_weff(const float* __restrict__ w, const float* __restrict__ cb,
                       const float* __restrict__ mA, const float* __restrict__ sA,
                       const float* __restrict__ mB, const float* __restrict__ sB,
                       _Float16* __restrict__ Weff, float* __restrict__ beff) {
  int b = blockIdx.y, o = blockIdx.x;
  float dot = 0.f;
  for (int c = threadIdx.x; c < CC; c += 256) {
    float w1 = w[o * 2 * CC + c], w2 = w[o * 2 * CC + CC + c];
    Weff[((size_t)b * CC + o) * CC + c] =
        (_Float16)(w1 * sA[b * CC + c] + w2 * sB[b * CC + c]);
    dot += w1 * mA[b * CC + c] + w2 * mB[b * CC + c];
  }
  __shared__ float red[256];
  red[threadIdx.x] = dot;
  __syncthreads();
  for (int s = 128; s > 0; s >>= 1) {
    if (threadIdx.x < s) red[threadIdx.x] += red[threadIdx.x + s];
    __syncthreads();
  }
  if (threadIdx.x == 0) beff[b * CC + o] = red[0] + cb[o];
}

// SAP logits: one wave per (b,n)
__global__ void k_saplogit(const float* __restrict__ x, const float* __restrict__ wv,
                           const float* __restrict__ bias, float* __restrict__ lg) {
  int idx = blockIdx.x * 8 + (threadIdx.x >> 5);
  int lane = threadIdx.x & 31;
  const float* p = x + (size_t)idx * CC;
  float d = 0.f;
  for (int c = lane; c < CC; c += 32) d += p[c] * wv[c];
#pragma unroll
  for (int o = 16; o > 0; o >>= 1) d += __shfl_down(d, o);
  if (lane == 0) lg[idx] = d + bias[0];
}

__global__ void k_softmax(float* lg) {
  float* p = lg + (size_t)blockIdx.x * SS;
  __shared__ float red[256];
  float m = -3.0e38f;
  for (int i = threadIdx.x; i < SS; i += 256) m = fmaxf(m, p[i]);
  red[threadIdx.x] = m;
  __syncthreads();
  for (int s = 128; s > 0; s >>= 1) {
    if (threadIdx.x < s) red[threadIdx.x] = fmaxf(red[threadIdx.x], red[threadIdx.x + s]);
    __syncthreads();
  }
  m = red[0];
  __syncthreads();
  float sum = 0.f;
  for (int i = threadIdx.x; i < SS; i += 256) {
    float e = __expf(p[i] - m);
    p[i] = e;
    sum += e;
  }
  red[threadIdx.x] = sum;
  __syncthreads();
  for (int s = 128; s > 0; s >>= 1) {
    if (threadIdx.x < s) red[threadIdx.x] += red[threadIdx.x + s];
    __syncthreads();
  }
  float inv = 1.f / red[0];
  for (int i = threadIdx.x; i < SS; i += 256) p[i] *= inv;
}

__global__ void k_sappool(const float* __restrict__ x, const float* __restrict__ a,
                          float* __restrict__ out) {
  int i = blockIdx.x * 256 + threadIdx.x;  // b*CC + c
  int b = i / CC, c = i % CC;
  const float* p = x + (size_t)b * SS * CC + c;
  const float* ap = a + (size_t)b * SS;
  float s = 0.f;
  for (int n = 0; n < SS; ++n) s += p[(size_t)n * CC] * ap[n];
  out[i] = s;
}

// out[b,c,t] = inorm(x)[b,c,t]*std_g[b,c] + mean_g[b,c] ; x is [B,T,C] f32
__global__ void k_final(const float* __restrict__ x, const float* __restrict__ xm,
                        const float* __restrict__ xsv, const float* __restrict__ stdg,
                        const float* __restrict__ meang, float* __restrict__ out) {
  int b = blockIdx.z;
  int t0 = blockIdx.x * 32, c0 = blockIdx.y * 32;
  __shared__ float tile[32][33];
#pragma unroll
  for (int j = 0; j < 4; ++j)
    tile[threadIdx.y + 8 * j][threadIdx.x] =
        x[((size_t)b * TT + t0 + threadIdx.y + 8 * j) * CC + c0 + threadIdx.x];
  __syncthreads();
#pragma unroll
  for (int j = 0; j < 4; ++j) {
    int c = c0 + threadIdx.y + 8 * j;
    float sc = xsv[b * CC + c] * stdg[b * CC + c];
    float sh = meang[b * CC + c] - xm[b * CC + c] * sc;
    out[((size_t)b * CC + c) * TT + t0 + threadIdx.x] =
        tile[threadIdx.x][threadIdx.y + 8 * j] * sc + sh;
  }
}

// ---------------------------------------------------------------------------
extern "C" void kernel_launch(void* const* d_in, const int* in_sizes, int n_in,
                              void* d_out, int out_size, void* d_ws, size_t ws_size,
                              hipStream_t stream) {
  (void)in_sizes; (void)n_in; (void)out_size; (void)ws_size;
  const float* src     = (const float*)d_in[0];
  const float* trg     = (const float*)d_in[1];
  const float* means   = (const float*)d_in[2];
  const float* stds    = (const float*)d_in[3];
  const float* wq_t    = (const float*)d_in[4];
  const float* wk_t    = (const float*)d_in[5];
  const float* wv_t    = (const float*)d_in[6];
  const float* wq_c    = (const float*)d_in[7];
  const float* wk_c    = (const float*)d_in[8];
  const float* wv_c    = (const float*)d_in[9];
  const float* conv1_w = (const float*)d_in[10];
  const float* conv1_b = (const float*)d_in[11];
  const float* conv2_w = (const float*)d_in[12];
  const float* bn_g    = (const float*)d_in[13];
  const float* bn_b    = (const float*)d_in[14];
  const float* bn_m    = (const float*)d_in[15];
  const float* bn_v    = (const float*)d_in[16];
  const float* msap_w  = (const float*)d_in[17];
  const float* msap_b  = (const float*)d_in[18];
  const float* ssap_w  = (const float*)d_in[19];
  const float* ssap_b  = (const float*)d_in[20];
  float* out = (float*)d_out;

  const size_t BTC = (size_t)BB * TT * CC;
  const size_t BC = (size_t)BB * CC;
  const size_t BT = (size_t)BB * TT;

  // carve workspace (~140 MB total)
  char* p = (char*)d_ws;
  auto alloc = [&](size_t bytes) {
    char* r = p;
    p += (bytes + 255) & ~(size_t)255;
    return r;
  };
  _Float16* Xt_src = (_Float16*)alloc(BTC * 2);
  _Float16* Xi_src = (_Float16*)alloc(BTC * 2);
  _Float16* Xt_trg = (_Float16*)alloc(BTC * 2);
  _Float16* Xi_trg = (_Float16*)alloc(BTC * 2);
  _Float16* Ytrg   = (_Float16*)alloc(BTC * 2);
  _Float16* q_t = (_Float16*)alloc(BTC * 2);
  _Float16* k_t = (_Float16*)alloc(BTC * 2);
  _Float16* v_t = (_Float16*)alloc(BTC * 2);
  _Float16* q_c = (_Float16*)alloc(BTC * 2);
  _Float16* k_c = (_Float16*)alloc(BTC * 2);
  _Float16* v_c = (_Float16*)alloc(BTC * 2);
  _Float16* vT_t = (_Float16*)alloc(BTC * 2);
  _Float16* vT_c = (_Float16*)alloc(BTC * 2);
  _Float16* wqt_h = (_Float16*)alloc((size_t)CC * CC * 2);
  _Float16* wkt_h = (_Float16*)alloc((size_t)CC * CC * 2);
  _Float16* wvt_h = (_Float16*)alloc((size_t)CC * CC * 2);
  _Float16* wqc_h = (_Float16*)alloc((size_t)CC * CC * 2);
  _Float16* wkc_h = (_Float16*)alloc((size_t)CC * CC * 2);
  _Float16* wvc_h = (_Float16*)alloc((size_t)CC * CC * 2);
  _Float16* w3h   = (_Float16*)alloc((size_t)3 * CC * CC * 2);
  _Float16* Weff  = (_Float16*)alloc(BC * CC * 2);
  _Float16* y1    = (_Float16*)alloc((size_t)BB * (TT + 2) * CC * 2);
  float* x      = (float*)alloc(BTC * 4);
  float* tm_src = (float*)alloc(BT * 4);
  float* tv_src = (float*)alloc(BT * 4);
  float* tm_trg = (float*)alloc(BT * 4);
  float* tv_trg = (float*)alloc(BT * 4);
  float* im_src = (float*)alloc(BC * 4);
  float* iv_src = (float*)alloc(BC * 4);
  float* im_trg = (float*)alloc(BC * 4);
  float* iv_trg = (float*)alloc(BC * 4);
  float* Gstats = (float*)alloc(4 * BC * 4);  // GmA,GvA,GmB,GvB contiguous
  float* GmA = Gstats, *GvA = Gstats + BC, *GmB = Gstats + 2 * BC, *GvB = Gstats + 3 * BC;
  float* beff   = (float*)alloc(BC * 4);
  float* bnsc   = (float*)alloc(CC * 4);
  float* bnsh   = (float*)alloc(CC * 4);
  float* xm     = (float*)alloc(BC * 4);
  float* xsv    = (float*)alloc(BC * 4);
  float* lg_m   = (float*)alloc((size_t)BB * SS * 4);
  float* lg_s   = (float*)alloc((size_t)BB * SS * 4);
  float* mean_g = (float*)alloc(BC * 4);
  float* std_g  = (float*)alloc(BC * 4);

  const float qscale = 1.f / sqrtf((float)CC);  // fold softmax temperature into Wq
  const int NW = CC * CC;

  // weight conversions
  k_f32f16<<<(NW + 255) / 256, 256, 0, stream>>>(wq_t, wqt_h, NW, qscale);
  k_f32f16<<<(NW + 255) / 256, 256, 0, stream>>>(wk_t, wkt_h, NW, 1.f);
  k_f32f16<<<(NW + 255) / 256, 256, 0, stream>>>(wv_t, wvt_h, NW, 1.f);
  k_f32f16<<<(NW + 255) / 256, 256, 0, stream>>>(wq_c, wqc_h, NW, qscale);
  k_f32f16<<<(NW + 255) / 256, 256, 0, stream>>>(wk_c, wkc_h, NW, 1.f);
  k_f32f16<<<(NW + 255) / 256, 256, 0, stream>>>(wv_c, wvc_h, NW, 1.f);
  k_w3<<<(NW + 255) / 256, 256, 0, stream>>>(conv2_w, w3h);
  k_bn<<<2, 256, 0, stream>>>(bn_g, bn_b, bn_m, bn_v, bnsc, bnsh);

  // normalization statistics
  k_tstats<<<dim3(TT / 64, BB), dim3(64, 4), 0, stream>>>(src, tm_src, tv_src);
  k_tstats<<<dim3(TT / 64, BB), dim3(64, 4), 0, stream>>>(trg, tm_trg, tv_trg);
  k_sstats<<<dim3(CC / 256, BB), 256, 0, stream>>>(src, im_src, iv_src,
                                                   (long long)CC * TT, TT, 1, CC, TT);
  k_sstats<<<dim3(CC / 256, BB), 256, 0, stream>>>(trg, im_trg, iv_trg,
                                                   (long long)CC * TT, TT, 1, CC, TT);

  // normalized + transposed f16 activations
  k_normxpose<<<dim3(TT / 32, CC / 32, BB), dim3(32, 8), 0, stream>>>(
      src, tm_src, tv_src, im_src, iv_src, Xt_src, Xi_src, (_Float16*)nullptr);
  k_normxpose<<<dim3(TT / 32, CC / 32, BB), dim3(32, 8), 0, stream>>>(
      trg, tm_trg, tv_trg, im_trg, iv_trg, Xt_trg, Xi_trg, Ytrg);

  // six projection GEMMs (WMMA, async-LDS staged)
  dim3 gg(TT / 64, CC / 128, BB);
  const long long SA = (long long)TT * CC;
  k_gemm<<<gg, 256, 0, stream>>>(Xt_src, SA, CC, wqt_h, 0, nullptr, 0, q_t, SA, CC, CC);
  k_gemm<<<gg, 256, 0, stream>>>(Xt_trg, SA, CC, wkt_h, 0, nullptr, 0, k_t, SA, CC, CC);
  k_gemm<<<gg, 256, 0, stream>>>(Ytrg,   SA, CC, wvt_h, 0, nullptr, 0, v_t, SA, CC, CC);
  k_gemm<<<gg, 256, 0, stream>>>(Xi_src, SA, CC, wqc_h, 0, nullptr, 0, q_c, SA, CC, CC);
  k_gemm<<<gg, 256, 0, stream>>>(Xi_trg, SA, CC, wkc_h, 0, nullptr, 0, k_c, SA, CC, CC);
  k_gemm<<<gg, 256, 0, stream>>>(Ytrg,   SA, CC, wvc_h, 0, nullptr, 0, v_c, SA, CC, CC);

  // V transposed for attention B-operand
  k_xpose16<<<dim3(TT / 32, CC / 32, BB), dim3(32, 8), 0, stream>>>(v_t, vT_t);
  k_xpose16<<<dim3(TT / 32, CC / 32, BB), dim3(32, 8), 0, stream>>>(v_c, vT_c);

  // attention (fused mean/var statistics)
  hipMemsetAsync(Gstats, 0, 4 * BC * 4, stream);
  k_attn<<<dim3(TT / 32, BB), 256, 0, stream>>>(q_t, k_t, vT_t, GmA, GvA);
  k_attn<<<dim3(TT / 32, BB), 256, 0, stream>>>(q_c, k_c, vT_c, GmB, GvB);
  k_statfin<<<(int)((BC + 255) / 256), 256, 0, stream>>>(GmA, GvA, (int)BC, 1.f / TT);
  k_statfin<<<(int)((BC + 255) / 256), 256, 0, stream>>>(GmB, GvB, (int)BC, 1.f / TT);

  // fold adaptive-norm affine + concat into conv1: effective per-batch weight
  k_weff<<<dim3(CC, BB), 256, 0, stream>>>(conv1_w, conv1_b, GmA, GvA, GmB, GvB,
                                           Weff, beff);
  hipMemsetAsync(y1, 0, (size_t)BB * (TT + 2) * CC * 2, stream);  // zero pad rows
  k_gemm<<<gg, 256, 0, stream>>>(Xi_src, SA, CC, Weff, (long long)CC * CC, beff, CC,
                                 y1 + CC, (long long)(TT + 2) * CC, CC, CC);

  // conv2 + BN + ReLU (WMMA, 3 shifted passes)
  k_conv2<<<gg, 256, 0, stream>>>(y1, w3h, bnsc, bnsh, x);

  // inorm stats of x over time
  k_sstats<<<dim3(CC / 256, BB), 256, 0, stream>>>(x, xm, xsv, (long long)TT * CC, 1,
                                                   CC, CC, TT);

  // GLAN: SAP pooling of means/stds
  k_saplogit<<<BB * SS / 8, 256, 0, stream>>>(means, msap_w, msap_b, lg_m);
  k_saplogit<<<BB * SS / 8, 256, 0, stream>>>(stds, ssap_w, ssap_b, lg_s);
  k_softmax<<<BB, 256, 0, stream>>>(lg_m);
  k_softmax<<<BB, 256, 0, stream>>>(lg_s);
  k_sappool<<<(int)(BC / 256), 256, 0, stream>>>(means, lg_m, mean_g);
  k_sappool<<<(int)(BC / 256), 256, 0, stream>>>(stds, lg_s, std_g);

  // final: inorm(x)*std + mean, transposed to [B,C,T]
  k_final<<<dim3(TT / 32, CC / 32, BB), dim3(32, 8), 0, stream>>>(x, xm, xsv, std_g,
                                                                  mean_g, out);
}
